// MultiAtomRetention_88132728914059
// MI455X (gfx1250) — matrix operand order repaired and checked by reading, not compiled, not run on hardware
//
#include <hip/hip_runtime.h>
#include <hip/hip_bf16.h>

#define N_ATOMS 4096
#define HID     1024
#define NH      8
#define DH      128
#define SCORE_SCALE 0.9971977f   // GAMMA ** (D^-0.5) = 0.96875 ** 0.08838835
#define LN_EPS  1e-5f
#define SSTR    36               // padded LDS stride for 16x32 score tiles

typedef __bf16 bf16_t;
typedef __attribute__((ext_vector_type(16))) __bf16 v16bf;
typedef __attribute__((ext_vector_type(8)))  float  v8f;
typedef __attribute__((ext_vector_type(4)))  unsigned u32x4;

union FragAB { v16bf v; u32x4 q[2]; };
union FragC  { v8f  v; float f[8]; };

static __device__ __forceinline__ v8f wmma_bf16(v16bf a, v16bf b, v8f c) {
    return __builtin_amdgcn_wmma_f32_16x16x32_bf16(
        false, a, false, b, (short)0, c, false, false);
}

static __device__ __forceinline__ bf16_t f2bf(float f) {
    unsigned u = __builtin_bit_cast(unsigned, f);
    u = u + 0x7FFFu + ((u >> 16) & 1u);           // round-to-nearest-even
    unsigned short s = (unsigned short)(u >> 16);
    return __builtin_bit_cast(bf16_t, s);
}

// A-fragment (16xK tile, K-chunk of 32) from row-major bf16, leading dim ld.
// ISA layout: lane&15 = M; lanes>=16 shift K by 8; elem e -> K = kb + (e>>3)*16 + (e&7)
static __device__ __forceinline__ v16bf load_a_rm(const bf16_t* base, int i0, int ld, int k0) {
    int lane = threadIdx.x & 31;
    int m  = lane & 15;
    int kb = (lane >> 4) << 3;                    // 0 or 8
    const bf16_t* p = base + (size_t)(i0 + m) * ld + k0 + kb;
    FragAB f;
    f.q[0] = *(const u32x4*)(p);                  // K = kb+0..7
    f.q[1] = *(const u32x4*)(p + 16);             // K = kb+16..23
    return f.v;
}

// B-fragment (Kx16 tile, K-chunk of 32) where column n is contiguous along K
// with leading dim ldk.  ISA layout: lane&15 = N; lanes>=16 shift K by 8; elem e -> K = kb + (e>>3)*16 + (e&7)
static __device__ __forceinline__ v16bf load_b_ck(const bf16_t* base, int n0, int ldk, int k0) {
    int lane = threadIdx.x & 31;
    int n  = lane & 15;
    int kb = (lane >> 4) << 3;                    // 0 or 8
    const bf16_t* p = base + (size_t)(n0 + n) * ldk + k0 + kb;
    FragAB f;
    f.q[0] = *(const u32x4*)(p);                  // K = kb+0..7
    f.q[1] = *(const u32x4*)(p + 16);             // K = kb+16..23
    return f.v;
}

// A-fragment from a 16x32 f32 tile living in LDS (row stride = stride f32),
// converted to bf16 on the fly.
static __device__ __forceinline__ v16bf load_a_lds_f32(const float* base, int stride) {
    int lane = threadIdx.x & 31;
    int m  = lane & 15;
    int kb = (lane >> 4) << 3;
    const float* p = base + m * stride + kb;
    v16bf r;
#pragma unroll
    for (int e = 0; e < 8; ++e) r[e] = f2bf(p[e]);
#pragma unroll
    for (int e = 0; e < 8; ++e) r[8 + e] = f2bf(p[16 + e]);
    return r;
}

// ---------------------------------------------------------------------------
// 1) prep: X -> bf16 row-major; W{q,k,v,o} -> bf16 transposed (col-major = B-frag friendly)
__global__ void prep_kernel(const float* __restrict__ X,
                            const float* __restrict__ Wq, const float* __restrict__ Wk,
                            const float* __restrict__ Wv, const float* __restrict__ Wo,
                            bf16_t* __restrict__ Xb,
                            bf16_t* __restrict__ WqT, bf16_t* __restrict__ WkT,
                            bf16_t* __restrict__ WvT, bf16_t* __restrict__ WoT) {
    int idx = blockIdx.x * blockDim.x + threadIdx.x;
    if (idx < N_ATOMS * HID) Xb[idx] = f2bf(X[idx]);
    if (idx < HID * HID) {
        int i = idx >> 10, j = idx & (HID - 1);
        int t = j * HID + i;                      // WT[out_col][in_k]
        WqT[t] = f2bf(Wq[idx]);
        WkT[t] = f2bf(Wk[idx]);
        WvT[t] = f2bf(Wv[idx]);
        WoT[t] = f2bf(Wo[idx]);
    }
}

// ---------------------------------------------------------------------------
// 2) row-wise softmax stats (max, 1/sum(exp)) for f_dist (y=0) and f_clb (y=1)
__global__ void rowstats_kernel(const float* __restrict__ dist, const float* __restrict__ clb,
                                float* __restrict__ dmax, float* __restrict__ dinv,
                                float* __restrict__ cmax, float* __restrict__ cinv) {
    const float* src = blockIdx.y ? clb : dist;
    float* omax = blockIdx.y ? cmax : dmax;
    float* oinv = blockIdx.y ? cinv : dinv;
    int row = blockIdx.x;
    const float* p = src + (size_t)row * N_ATOMS;
    float vals[16];
    float mx = -INFINITY;
#pragma unroll
    for (int s = 0; s < 16; ++s) { vals[s] = p[threadIdx.x + 256 * s]; mx = fmaxf(mx, vals[s]); }
    __shared__ float red[256];
    red[threadIdx.x] = mx; __syncthreads();
    for (int off = 128; off; off >>= 1) {
        if (threadIdx.x < off) red[threadIdx.x] = fmaxf(red[threadIdx.x], red[threadIdx.x + off]);
        __syncthreads();
    }
    mx = red[0]; __syncthreads();
    float sm = 0.f;
#pragma unroll
    for (int s = 0; s < 16; ++s) sm += __expf(vals[s] - mx);
    red[threadIdx.x] = sm; __syncthreads();
    for (int off = 128; off; off >>= 1) {
        if (threadIdx.x < off) red[threadIdx.x] += red[threadIdx.x + off];
        __syncthreads();
    }
    if (threadIdx.x == 0) { omax[row] = mx; oinv[row] = 1.0f / red[0]; }
}

// ---------------------------------------------------------------------------
// 3) projections: Q,K per-head row-major [h][n][128]; V per-head transposed [h][128][n]
__global__ void __launch_bounds__(256) proj_kernel(
        const bf16_t* __restrict__ Xb,
        const bf16_t* __restrict__ WqT, const bf16_t* __restrict__ WkT,
        const bf16_t* __restrict__ WvT,
        bf16_t* __restrict__ Qh, bf16_t* __restrict__ Kh, bf16_t* __restrict__ Vt) {
    int i0  = blockIdx.x * 16;
    int w   = threadIdx.x >> 5;
    int jt  = blockIdx.y * 8 + w;                 // output column tile 0..63
    int mat = blockIdx.z;                         // 0=Q 1=K 2=V (uniform per block)
    const bf16_t* WT = (mat == 0) ? WqT : (mat == 1) ? WkT : WvT;
    v8f acc = {};
    for (int k0 = 0; k0 < HID; k0 += 32) {
        v16bf a = load_a_rm(Xb, i0, HID, k0);
        v16bf b = load_b_ck(WT, jt * 16, HID, k0);
        acc = wmma_bf16(a, b, acc);
    }
    int lane = threadIdx.x & 31;
    int nl = lane & 15;
    int mbase = (lane >> 4) << 3;
    FragC c; c.v = acc;
#pragma unroll
    for (int r = 0; r < 8; ++r) {
        int i = i0 + mbase + r;
        int col = jt * 16 + nl;
        int h = col >> 7, d = col & (DH - 1);
        bf16_t val = f2bf(c.f[r]);
        if (mat == 0)      Qh[(size_t)h * (N_ATOMS * DH) + (size_t)i * DH + d] = val;
        else if (mat == 1) Kh[(size_t)h * (N_ATOMS * DH) + (size_t)i * DH + d] = val;
        else               Vt[(size_t)h * (DH * N_ATOMS) + (size_t)d * N_ATOMS + i] = val;
    }
}

// ---------------------------------------------------------------------------
// 4) retention: workgroup = 16 query rows, wave w = head w.
__global__ void __launch_bounds__(256) retention_kernel(
        const bf16_t* __restrict__ Qh, const bf16_t* __restrict__ Kh,
        const bf16_t* __restrict__ Vt,
        const float* __restrict__ f_adj, const float* __restrict__ f_dist,
        const float* __restrict__ f_clb,
        const float* __restrict__ dmax, const float* __restrict__ dinv,
        const float* __restrict__ cmax, const float* __restrict__ cinv,
        float* __restrict__ mean_out, bf16_t* __restrict__ Retb) {
    __shared__ float S_lds[NH][16][SSTR];
    int i0   = blockIdx.x * 16;
    int h    = threadIdx.x >> 5;
    int lane = threadIdx.x & 31;
    int nl   = lane & 15;
    int mbase = (lane >> 4) << 3;

    const bf16_t* q  = Qh + (size_t)h * (N_ATOMS * DH);
    const bf16_t* k  = Kh + (size_t)h * (N_ATOMS * DH);
    const bf16_t* vt = Vt + (size_t)h * (DH * N_ATOMS);

    v16bf qa[4];
#pragma unroll
    for (int c = 0; c < 4; ++c) qa[c] = load_a_rm(q, i0, DH, c * 32);

    const float* Bm = (h < 2) ? f_adj : (h < 4) ? f_dist : f_clb;
    float rmx[8], rnv[8];
#pragma unroll
    for (int r = 0; r < 8; ++r) { rmx[r] = 0.f; rnv[r] = 0.f; }
    if (h >= 2 && h < 6) {
        const float* mx = (h < 4) ? dmax : cmax;
        const float* nv = (h < 4) ? dinv : cinv;
#pragma unroll
        for (int r = 0; r < 8; ++r) { rmx[r] = mx[i0 + mbase + r]; rnv[r] = nv[i0 + mbase + r]; }
    }

    v8f ret[8];
    v8f vzero = {};
#pragma unroll
    for (int c = 0; c < 8; ++c) ret[c] = vzero;

    for (int j0 = 0; j0 < N_ATOMS; j0 += 32) {
        v8f s0 = vzero, s1 = vzero;
#pragma unroll
        for (int c = 0; c < 4; ++c) {
            v16bf b0 = load_b_ck(k, j0,      DH, c * 32);
            s0 = wmma_bf16(qa[c], b0, s0);
            v16bf b1 = load_b_ck(k, j0 + 16, DH, c * 32);
            s1 = wmma_bf16(qa[c], b1, s1);
        }
        FragC c0, c1; c0.v = s0; c1.v = s1;
#pragma unroll
        for (int r = 0; r < 8; ++r) {
            int row = i0 + mbase + r;
            float v0 = c0.f[r], v1 = c1.f[r];
            if (h < 6) {
                float b0 = Bm[(size_t)row * N_ATOMS + j0 + nl];
                float b1 = Bm[(size_t)row * N_ATOMS + j0 + 16 + nl];
                if (h >= 2) {
                    b0 = __expf(b0 - rmx[r]) * rnv[r];
                    b1 = __expf(b1 - rmx[r]) * rnv[r];
                }
                v0 += b0; v1 += b1;
            }
            v0 *= SCORE_SCALE; v1 *= SCORE_SCALE;
            S_lds[h][mbase + r][nl]      = v0;
            S_lds[h][mbase + r][16 + nl] = v1;
        }
        __syncthreads();
        // cross-head mean -> d_out score region
#pragma unroll
        for (int rep = 0; rep < 2; ++rep) {
            int idx = threadIdx.x + rep * 256;
            int M = idx >> 5, n = idx & 31;
            float s = 0.f;
#pragma unroll
            for (int hh = 0; hh < NH; ++hh) s += S_lds[hh][M][n];
            mean_out[(size_t)(i0 + M) * N_ATOMS + j0 + n] = s * 0.125f;
        }
        // ret += S(16x32) @ V(32x128)
        v16bf sa = load_a_lds_f32(&S_lds[h][0][0], SSTR);
#pragma unroll
        for (int c = 0; c < 8; ++c) {
            v16bf vb = load_b_ck(vt, c * 16, N_ATOMS, j0);
            ret[c] = wmma_bf16(sa, vb, ret[c]);
        }
        __syncthreads();
    }
    // ReLU + store ret (bf16, row-major [N][HID])
#pragma unroll
    for (int c = 0; c < 8; ++c) {
        FragC rc; rc.v = ret[c];
#pragma unroll
        for (int r = 0; r < 8; ++r) {
            float v = fmaxf(rc.f[r], 0.f);
            int row = i0 + mbase + r;
            int col = h * DH + c * 16 + nl;
            Retb[(size_t)row * HID + col] = f2bf(v);
        }
    }
}

// ---------------------------------------------------------------------------
// 5) output projection + residual + LayerNorm
__global__ void __launch_bounds__(256) outln_kernel(
        const bf16_t* __restrict__ Retb, const bf16_t* __restrict__ WoT,
        const float* __restrict__ X, const float* __restrict__ bo,
        const float* __restrict__ ln_g, const float* __restrict__ ln_b,
        float* __restrict__ out) {
    __shared__ float part[NH][16];
    __shared__ float mu_s[16];
    __shared__ float rstd_s[16];
    int i0   = blockIdx.x * 16;
    int w    = threadIdx.x >> 5;
    int lane = threadIdx.x & 31;
    int nl   = lane & 15;
    int mbase = (lane >> 4) << 3;

    v8f acc[8];
    v8f vzero = {};
#pragma unroll
    for (int t = 0; t < 8; ++t) acc[t] = vzero;
    for (int k0 = 0; k0 < HID; k0 += 32) {
        v16bf a = load_a_rm(Retb, i0, HID, k0);
#pragma unroll
        for (int t = 0; t < 8; ++t) {
            v16bf b = load_b_ck(WoT, (w * 8 + t) * 16, HID, k0);
            acc[t] = wmma_bf16(a, b, acc[t]);
        }
    }
    float y[8][8];
    float prow[8];
#pragma unroll
    for (int r = 0; r < 8; ++r) prow[r] = 0.f;
#pragma unroll
    for (int t = 0; t < 8; ++t) {
        FragC c; c.v = acc[t];
        int col = (w * 8 + t) * 16 + nl;
#pragma unroll
        for (int r = 0; r < 8; ++r) {
            int row = i0 + mbase + r;
            float v = c.f[r] + bo[col] + X[(size_t)row * HID + col];
            y[t][r] = v;
            prow[r] += v;
        }
    }
#pragma unroll
    for (int off = 1; off < 16; off <<= 1)
#pragma unroll
        for (int r = 0; r < 8; ++r) prow[r] += __shfl_xor(prow[r], off, 32);
    if (nl == 0)
#pragma unroll
        for (int r = 0; r < 8; ++r) part[w][mbase + r] = prow[r];
    __syncthreads();
    if (threadIdx.x < 16) {
        float s = 0.f;
#pragma unroll
        for (int hh = 0; hh < NH; ++hh) s += part[hh][threadIdx.x];
        mu_s[threadIdx.x] = s * (1.0f / HID);
    }
    __syncthreads();
#pragma unroll
    for (int r = 0; r < 8; ++r) prow[r] = 0.f;
#pragma unroll
    for (int t = 0; t < 8; ++t)
#pragma unroll
        for (int r = 0; r < 8; ++r) {
            float d = y[t][r] - mu_s[mbase + r];
            prow[r] += d * d;
        }
#pragma unroll
    for (int off = 1; off < 16; off <<= 1)
#pragma unroll
        for (int r = 0; r < 8; ++r) prow[r] += __shfl_xor(prow[r], off, 32);
    if (nl == 0)
#pragma unroll
        for (int r = 0; r < 8; ++r) part[w][mbase + r] = prow[r];
    __syncthreads();
    if (threadIdx.x < 16) {
        float s = 0.f;
#pragma unroll
        for (int hh = 0; hh < NH; ++hh) s += part[hh][threadIdx.x];
        rstd_s[threadIdx.x] = rsqrtf(s * (1.0f / HID) + LN_EPS);
    }
    __syncthreads();
#pragma unroll
    for (int t = 0; t < 8; ++t) {
        int col = (w * 8 + t) * 16 + nl;
        float g = ln_g[col], bb = ln_b[col];
#pragma unroll
        for (int r = 0; r < 8; ++r) {
            int row = i0 + mbase + r;
            out[(size_t)row * HID + col] =
                (y[t][r] - mu_s[mbase + r]) * rstd_s[mbase + r] * g + bb;
        }
    }
}

// ---------------------------------------------------------------------------
extern "C" void kernel_launch(void* const* d_in, const int* in_sizes, int n_in,
                              void* d_out, int out_size, void* d_ws, size_t ws_size,
                              hipStream_t stream) {
    (void)in_sizes; (void)n_in; (void)out_size; (void)ws_size;
    const float* X      = (const float*)d_in[0];
    const float* f_adj  = (const float*)d_in[1];
    const float* f_dist = (const float*)d_in[2];
    const float* f_clb  = (const float*)d_in[3];
    const float* Wq     = (const float*)d_in[4];
    const float* Wk     = (const float*)d_in[5];
    const float* Wv     = (const float*)d_in[6];
    const float* Wo     = (const float*)d_in[7];
    const float* bo     = (const float*)d_in[8];
    const float* ln_g   = (const float*)d_in[9];
    const float* ln_b   = (const float*)d_in[10];
    float* out = (float*)d_out;

    char* ws = (char*)d_ws;
    size_t off = 0;
    bf16_t* Xb   = (bf16_t*)(ws + off); off += (size_t)N_ATOMS * HID * 2;
    bf16_t* WqT  = (bf16_t*)(ws + off); off += (size_t)HID * HID * 2;
    bf16_t* WkT  = (bf16_t*)(ws + off); off += (size_t)HID * HID * 2;
    bf16_t* WvT  = (bf16_t*)(ws + off); off += (size_t)HID * HID * 2;
    bf16_t* WoT  = (bf16_t*)(ws + off); off += (size_t)HID * HID * 2;
    bf16_t* Qh   = (bf16_t*)(ws + off); off += (size_t)N_ATOMS * HID * 2;
    bf16_t* Kh   = (bf16_t*)(ws + off); off += (size_t)N_ATOMS * HID * 2;
    bf16_t* Vt   = (bf16_t*)(ws + off); off += (size_t)N_ATOMS * HID * 2;
    bf16_t* Retb = (bf16_t*)(ws + off); off += (size_t)N_ATOMS * HID * 2;
    float* dmax  = (float*)(ws + off); off += (size_t)N_ATOMS * 4;
    float* dinv  = (float*)(ws + off); off += (size_t)N_ATOMS * 4;
    float* cmax  = (float*)(ws + off); off += (size_t)N_ATOMS * 4;
    float* cinv  = (float*)(ws + off); off += (size_t)N_ATOMS * 4;

    prep_kernel<<<(N_ATOMS * HID) / 256, 256, 0, stream>>>(
        X, Wq, Wk, Wv, Wo, Xb, WqT, WkT, WvT, WoT);
    rowstats_kernel<<<dim3(N_ATOMS, 2), 256, 0, stream>>>(
        f_dist, f_clb, dmax, dinv, cmax, cinv);
    proj_kernel<<<dim3(N_ATOMS / 16, 8, 3), 256, 0, stream>>>(
        Xb, WqT, WkT, WvT, Qh, Kh, Vt);
    retention_kernel<<<N_ATOMS / 16, 256, 0, stream>>>(
        Qh, Kh, Vt, f_adj, f_dist, f_clb, dmax, dinv, cmax, cinv,
        out + (size_t)N_ATOMS * HID, Retb);
    outln_kernel<<<N_ATOMS / 16, 256, 0, stream>>>(
        Retb, WoT, X, bo, ln_g, ln_b, out);
}
